// SAE_13546326852276
// MI455X (gfx1250) — compile-verified, matrix operand order checked by reading, not run on hardware
//
#include <hip/hip_runtime.h>
#include <hip/hip_bf16.h>
#include <math.h>

// ---------------- problem constants ----------------
#define BN   2048   // batch
#define DK   512    // input dim
#define HN   2048   // hidden dim
#define NS   32     // samples
#define BETA 1.0f

typedef __bf16 v16bf __attribute__((ext_vector_type(16)));
typedef __bf16 v8bf  __attribute__((ext_vector_type(8)));
typedef float  v8f   __attribute__((ext_vector_type(8)));
typedef float  v4f   __attribute__((ext_vector_type(4)));

// output layout (flat f32, concatenated in return order)
static constexpr long long OFF_H  = 0;
static constexpr long long OFF_Z  = (long long)BN * HN;                  //   4,194,304
static constexpr long long OFF_OH = OFF_Z  + (long long)BN * HN;         //   8,388,608
static constexpr long long OFF_XD = OFF_OH + (long long)NS * BN * HN;    // 142,606,336
static constexpr long long OFF_WT = OFF_XD + (long long)NS * BN * DK;    // 176,160,768

// ---------------- 1a. x (f32) -> bf16 ----------------
__global__ void cvt_x_kernel(const float* __restrict__ x, __bf16* __restrict__ xb, int n) {
    int i = blockIdx.x * blockDim.x + threadIdx.x;
    if (i < n) xb[i] = (__bf16)x[i];
}

// ---------------- 1b. W [DK,HN] -> Wt [HN,DK] (f32 + bf16), LDS tiled ----------------
__global__ void transpose_w_kernel(const float* __restrict__ W,
                                   float* __restrict__ Wt, __bf16* __restrict__ Wtb) {
    __shared__ float tile[32][33];
    int hh0 = blockIdx.x * 32;   // column in W (hidden)
    int d0  = blockIdx.y * 32;   // row in W (input dim)
    int tx = threadIdx.x, ty = threadIdx.y;
    #pragma unroll
    for (int j = ty; j < 32; j += 8)
        tile[j][tx] = W[(long long)(d0 + j) * HN + hh0 + tx];   // coalesced over hh
    __syncthreads();
    #pragma unroll
    for (int j = ty; j < 32; j += 8) {
        float v = tile[tx][j];                                   // [d_local=tx][hh_local=j]
        long long o = (long long)(hh0 + j) * DK + d0 + tx;       // coalesced over d
        Wt[o]  = v;
        Wtb[o] = (__bf16)v;
    }
}

// ---------------- 2. h = x @ W via v_wmma_f32_16x16x32_bf16 ----------------
// one wave -> one 16x16 output tile; 8 waves/block stacked along M.
// A per-lane layout (16x32 bf16): lane<16 -> K {0..7,16..23}, lane>=16 -> K {8..15,24..31}
// B is read from Wt(bf16) so each column is a contiguous row -> same 16B-load pattern.
__global__ void gemm_h_kernel(const __bf16* __restrict__ xb,
                              const __bf16* __restrict__ wtb,
                              float* __restrict__ h) {
    int lane = threadIdx.x & 31;
    int wv   = threadIdx.x >> 5;
    int l16  = lane & 15;
    int half = lane >> 4;
    int mtile = blockIdx.y * 8 + wv;
    int ntile = blockIdx.x;

    const __bf16* arow = xb  + (long long)(mtile * 16 + l16) * DK;
    const __bf16* brow = wtb + (long long)(ntile * 16 + l16) * DK;
    int klo = half * 8;

    v8f c = {};
    for (int kb = 0; kb < DK; kb += 32) {
        if (kb + 32 < DK) {                       // CDNA5 global_prefetch hints
            __builtin_prefetch(arow + kb + 32, 0, 0);
            __builtin_prefetch(brow + kb + 32, 0, 0);
        }
        v8bf a0 = *(const v8bf*)(arow + kb + klo);
        v8bf a1 = *(const v8bf*)(arow + kb + 16 + klo);
        v8bf b0 = *(const v8bf*)(brow + kb + klo);
        v8bf b1 = *(const v8bf*)(brow + kb + 16 + klo);
        v16bf a, b;
        #pragma unroll
        for (int i = 0; i < 8; ++i) {
            a[i] = a0[i]; a[i + 8] = a1[i];
            b[i] = b0[i]; b[i + 8] = b1[i];
        }
        c = __builtin_amdgcn_wmma_f32_16x16x32_bf16(
                /*neg_a=*/false, a, /*neg_b=*/false, b,
                /*c_mod=*/(short)0, c, /*reuse_a=*/false, /*reuse_b=*/false);
    }
    int orow = mtile * 16 + half * 8;
    int ocol = ntile * 16 + l16;
    #pragma unroll
    for (int r = 0; r < 8; ++r)
        h[(long long)(orow + r) * HN + ocol] = c[r];
}

// ---------------- 3. z = softmax(h * BETA) rowwise ----------------
__global__ void softmax_kernel(const float* __restrict__ h, float* __restrict__ z) {
    __shared__ float red[256];
    int b = blockIdx.x;
    const float* hr = h + (long long)b * HN;
    float mx = -__builtin_inff();
    for (int i = threadIdx.x; i < HN; i += 256) mx = fmaxf(mx, hr[i] * BETA);
    red[threadIdx.x] = mx; __syncthreads();
    for (int s = 128; s > 0; s >>= 1) {
        if (threadIdx.x < s) red[threadIdx.x] = fmaxf(red[threadIdx.x], red[threadIdx.x + s]);
        __syncthreads();
    }
    mx = red[0]; __syncthreads();
    float sum = 0.f;
    for (int i = threadIdx.x; i < HN; i += 256) sum += __expf(hr[i] * BETA - mx);
    red[threadIdx.x] = sum; __syncthreads();
    for (int s = 128; s > 0; s >>= 1) {
        if (threadIdx.x < s) red[threadIdx.x] += red[threadIdx.x + s];
        __syncthreads();
    }
    float inv = 1.f / red[0];
    for (int i = threadIdx.x; i < HN; i += 256)
        z[(long long)b * HN + i] = __expf(hr[i] * BETA - mx) * inv;
}

// ---------------- 4. categorical stand-in: Gumbel-argmax over z-as-logits ----------------
__device__ __forceinline__ unsigned hash3(unsigned a, unsigned b, unsigned c) {
    unsigned h = a * 0x9E3779B1u ^ b * 0x85EBCA77u ^ c * 0xC2B2AE3Du;
    h ^= h >> 15; h *= 0x2C1B3C6Du; h ^= h >> 12; h *= 0x297A2D39u; h ^= h >> 15;
    return h;
}
__global__ void sample_kernel(const float* __restrict__ z, int* __restrict__ samp) {
    int gw   = (int)((blockIdx.x * blockDim.x + threadIdx.x) >> 5);  // 0..NS*BN-1
    int lane = threadIdx.x & 31;
    int n  = gw >> 11;          // / BN
    int bb = gw & (BN - 1);
    // reference reshape scramble: samp[n,bb] = orig[(n*BN+bb)%32, (n*BN+bb)/32]
    int row     = n * 64 + (bb >> 5);
    unsigned ns = (unsigned)(bb & 31);
    const float* zr = z + (long long)row * HN;
    float best = -__builtin_inff();
    int   bidx = 0;
    for (int hcol = lane; hcol < HN; hcol += 32) {
        unsigned r = hash3(0x2545F491u + ns, (unsigned)row, (unsigned)hcol);
        float u = (float)(r >> 8) * (1.0f / 16777216.0f) + 1e-12f;
        float g = -__logf(-__logf(u));
        float v = zr[hcol] + g;
        if (v > best) { best = v; bidx = hcol; }
    }
    #pragma unroll
    for (int off = 16; off > 0; off >>= 1) {
        float ov = __shfl_down(best, off, 32);
        int   oi = __shfl_down(bidx, off, 32);
        if (ov > best) { best = ov; bidx = oi; }
    }
    if (lane == 0) samp[gw] = bidx;
}

// ---------------- 5. onehot: stream-zero then scatter ones ----------------
__global__ void zero_kernel(float* __restrict__ p, long long n4) {
    long long i = (long long)blockIdx.x * blockDim.x + threadIdx.x;
    long long stride = (long long)gridDim.x * blockDim.x;
    v4f zv = {0.f, 0.f, 0.f, 0.f};
    for (; i < n4; i += stride) ((v4f*)p)[i] = zv;
}
__global__ void scatter_kernel(const int* __restrict__ samp, float* __restrict__ oh) {
    int i = blockIdx.x * blockDim.x + threadIdx.x;       // 0..NS*BN-1
    oh[(long long)i * HN + samp[i]] = 1.0f;
}

// ---------------- 6. gather decode + Gaussian kernel temp ----------------
__global__ void decode_kernel(const float* __restrict__ x, const float* __restrict__ Wt,
                              const int* __restrict__ samp,
                              float* __restrict__ xd, float* __restrict__ temp) {
    int gw   = (int)((blockIdx.x * blockDim.x + threadIdx.x) >> 5);  // (n,b) pair
    int lane = threadIdx.x & 31;
    int b = gw & (BN - 1);
    int idx = samp[gw];
    const v4f* wr = (const v4f*)(Wt + (long long)idx * DK);
    const v4f* xr = (const v4f*)(x  + (long long)b   * DK);
    v4f*       od = (v4f*)(xd + (long long)gw * DK);
    float s = 0.f;
    #pragma unroll
    for (int j = 0; j < 4; ++j) {
        int e = lane + j * 32;            // 128 float4s per row
        v4f w  = wr[e];
        v4f xv = xr[e];
        od[e] = w;
        v4f d = xv - w;
        s += d.x * d.x + d.y * d.y + d.z * d.z + d.w * d.w;
    }
    #pragma unroll
    for (int off = 16; off > 0; off >>= 1) s += __shfl_down(s, off, 32);
    if (lane == 0) temp[gw] = 0.3989422804014327f * __expf(-0.5f * s);
}

// ---------------- 7. weight = temp / mean_n(temp) ----------------
__global__ void weight_kernel(const float* __restrict__ temp, float* __restrict__ wout) {
    int b = blockIdx.x * blockDim.x + threadIdx.x;
    if (b >= BN) return;
    float s = 0.f;
    #pragma unroll
    for (int n = 0; n < NS; ++n) s += temp[n * BN + b];
    float invmean = (float)NS / s;
    #pragma unroll
    for (int n = 0; n < NS; ++n) wout[n * BN + b] = temp[n * BN + b] * invmean;
}

// ---------------- launch ----------------
extern "C" void kernel_launch(void* const* d_in, const int* in_sizes, int n_in,
                              void* d_out, int out_size, void* d_ws, size_t ws_size,
                              hipStream_t stream) {
    const float* x = (const float*)d_in[0];   // [2048, 512]
    const float* W = (const float*)d_in[1];   // [512, 2048]
    float* out = (float*)d_out;

    // workspace carve-up (~8.5 MB)
    char* ws = (char*)d_ws;
    float*  Wt   = (float*)(ws);                          // 4 MB  [HN, DK] f32
    __bf16* xb   = (__bf16*)(ws + (4u << 20));            // 2 MB  [BN, DK] bf16
    __bf16* wtb  = (__bf16*)(ws + (6u << 20));            // 2 MB  [HN, DK] bf16
    int*    samp = (int*)  (ws + (8u << 20));             // 256 KB [NS, BN]
    float*  temp = (float*)(ws + (8u << 20) + (256u << 10)); // 256 KB [NS, BN]

    float* h_out  = out + OFF_H;
    float* z_out  = out + OFF_Z;
    float* oh_out = out + OFF_OH;
    float* xd_out = out + OFF_XD;
    float* wt_out = out + OFF_WT;

    // 1. precision/layout prep
    cvt_x_kernel<<<(BN * DK) / 256, 256, 0, stream>>>(x, xb, BN * DK);
    transpose_w_kernel<<<dim3(HN / 32, DK / 32), dim3(32, 8), 0, stream>>>(W, Wt, wtb);

    // 2. WMMA GEMM -> h
    gemm_h_kernel<<<dim3(HN / 16, BN / 16 / 8), 256, 0, stream>>>(xb, wtb, h_out);

    // 3. softmax -> z
    softmax_kernel<<<BN, 256, 0, stream>>>(h_out, z_out);

    // 4. sampling (one wave per output element)
    sample_kernel<<<(NS * BN * 32) / 256, 256, 0, stream>>>(z_out, samp);

    // 5. onehot
    zero_kernel<<<32768, 256, 0, stream>>>(oh_out, (long long)NS * BN * HN / 4);
    scatter_kernel<<<(NS * BN) / 256, 256, 0, stream>>>(samp, oh_out);

    // 6. decode + gaussian temp (one wave per (n,b))
    decode_kernel<<<(NS * BN * 32) / 256, 256, 0, stream>>>(x, Wt, samp, xd_out, temp);

    // 7. weight
    weight_kernel<<<(BN + 255) / 256, 256, 0, stream>>>(temp, wt_out);
}
